// MultiViewCrossAttention_7043746365947
// MI455X (gfx1250) — compile-verified, matrix operand order checked by reading, not run on hardware
//
#include <hip/hip_runtime.h>
#include <hip/hip_bf16.h>

// ---------------------------------------------------------------------------
// MultiViewCrossAttention for MI455X (gfx1250, wave32, WMMA + async-to-LDS).
//
// pooled[b,c] = ( mean_pix(Xb)[b,c] + 3*bz[c]
//                 + sum_d Wz[c,d] * (sum_v mean_q Z_v[b,q,d]) ) / 4
// out = pooled @ Wfc^T + bfc
// => attention kernel only needs column sums over q of the attention output.
// ---------------------------------------------------------------------------

#define NV    4
#define NB    8
#define CH    128
#define C2    64
#define AREA  2304           // 48*48
#define NQT   144            // AREA/16 query tiles
#define NKIT  72             // AREA/32 key iterations (32 keys/iter)

typedef _Float16 v16h __attribute__((ext_vector_type(16)));
typedef _Float16 h8   __attribute__((ext_vector_type(8)));
typedef _Float16 h2   __attribute__((ext_vector_type(2)));
typedef float    v8f  __attribute__((ext_vector_type(8)));

// Build a 16-half WMMA operand from two contiguous 16-byte chunks.
// Documented A layout (16-bit 16x32, wave32): lane l (m=l&15, h=l>>4) holds
// row M=m, K = {8h..8h+7} in regs 0-3 (pairs) and K = {16+8h..16+8h+7} in
// regs 4-7.  B operand assumed mirrored (col N=l&15, same K split).
__device__ __forceinline__ v16h mv_frag(const _Float16* lo, const _Float16* hi) {
  h8 a = *(const h8*)lo;
  h8 b = *(const h8*)hi;
  return __builtin_shufflevector(a, b, 0,1,2,3,4,5,6,7,8,9,10,11,12,13,14,15);
}

__device__ __forceinline__ v8f mv_wmma(v16h a, v16h b, v8f c) {
  return __builtin_amdgcn_wmma_f32_16x16x32_f16(
      /*neg_a=*/false, a, /*neg_b=*/false, b,
      /*c_mod=*/(short)0, c, /*reuse_a=*/false, /*reuse_b=*/false);
}

// Butterfly step within each 16-lane row via v_permlane16_b32 (VALU path,
// no LDS traffic, co-executes with XDL WMMA).
template <unsigned S0, unsigned S1>
__device__ __forceinline__ unsigned mv_pl16(unsigned v) {
  return __builtin_amdgcn_permlane16(v, v, S0, S1, false, false);
}
__device__ __forceinline__ unsigned mv_hmax2(unsigned a, unsigned b) {
  h2 x = __builtin_bit_cast(h2, a);
  h2 y = __builtin_bit_cast(h2, b);
  h2 r = __builtin_elementwise_max(x, y);
  return __builtin_bit_cast(unsigned, r);
}

// Low 32 bits of a generic shared-memory address == workgroup LDS byte offset.
__device__ __forceinline__ unsigned mv_lds_off(const void* p) {
  return (unsigned)(unsigned long long)p;
}

// gfx1250 async copy global -> LDS, 16 bytes per lane (inline asm: the
// bridge doc's recommended portable path for the async/TDM family).
__device__ __forceinline__ void mv_async_ld16(unsigned lds_byte_off, const _Float16* g) {
  asm volatile("global_load_async_to_lds_b128 %0, %1, off"
               :: "v"(lds_byte_off), "v"((unsigned long long)g)
               : "memory");
}

// ---------------------------------------------------------------------------
// Stage 1a: Q projection.  Qh[b][q][d] f16, softmax scale 1/sqrt(64) folded in.
// ---------------------------------------------------------------------------
__global__ void __launch_bounds__(256)
mv_proj_q(const float* __restrict__ povs, const float* __restrict__ Wq,
          const float* __restrict__ bq, const int* __restrict__ basep,
          _Float16* __restrict__ Qh) {
  int idx = blockIdx.x * 256 + threadIdx.x;          // [b][pix][o], o fastest
  int o   = idx & 63;
  int pix = (idx >> 6) % AREA;
  int b   = idx / (AREA * 64);
  int base = *basep;
  const float* X = povs + ((size_t)base * NB + b) * (size_t)CH * AREA + pix;
  const float* w = Wq + o * CH;
  float acc = bq[o];
#pragma unroll 8
  for (int c = 0; c < CH; ++c) acc += X[(size_t)c * AREA] * w[c];
  Qh[idx] = (_Float16)(acc * 0.125f);                // 1/sqrt(c2)
}

// ---------------------------------------------------------------------------
// Stage 1b: K projection.  Kh[ov][b][key][d] f16 (key-major rows of 64 halves).
// ---------------------------------------------------------------------------
__global__ void __launch_bounds__(256)
mv_proj_k(const float* __restrict__ povs, const float* __restrict__ Wk,
          const float* __restrict__ bk, const int* __restrict__ basep,
          _Float16* __restrict__ Kh) {
  int idx = blockIdx.x * 256 + threadIdx.x;          // [ov][b][pix][o]
  int o   = idx & 63;
  int pix = (idx >> 6) % AREA;
  int vb  = idx / (AREA * 64);
  int b   = vb & 7;
  int ov  = vb >> 3;
  int base = *basep;
  int view = (ov < base) ? ov : ov + 1;
  const float* X = povs + ((size_t)view * NB + b) * (size_t)CH * AREA + pix;
  const float* w = Wk + o * CH;
  float acc = bk[o];
#pragma unroll 8
  for (int c = 0; c < CH; ++c) acc += X[(size_t)c * AREA] * w[c];
  Kh[idx] = (_Float16)acc;
}

// ---------------------------------------------------------------------------
// Stage 1c: V projection.  Vh[ov][b][d][key] f16 (d-major rows of AREA halves).
// ---------------------------------------------------------------------------
__global__ void __launch_bounds__(256)
mv_proj_v(const float* __restrict__ povs, const float* __restrict__ Wv,
          const float* __restrict__ bv, const int* __restrict__ basep,
          _Float16* __restrict__ Vh) {
  int idx = blockIdx.x * 256 + threadIdx.x;          // [ov][b][d][pix]
  int pix = idx % AREA;
  int d   = (idx / AREA) & 63;
  int vb  = idx / (AREA * 64);
  int b   = vb & 7;
  int ov  = vb >> 3;
  int base = *basep;
  int view = (ov < base) ? ov : ov + 1;
  const float* X = povs + ((size_t)view * NB + b) * (size_t)CH * AREA + pix;
  const float* w = Wv + d * CH;
  float acc = bv[d];
#pragma unroll 8
  for (int c = 0; c < CH; ++c) acc += X[(size_t)c * AREA] * w[c];
  Vh[idx] = (_Float16)acc;
}

// ---------------------------------------------------------------------------
// Stage 1d: spatial mean of base view: meanXb[b*128+c].  One wave per row.
// ---------------------------------------------------------------------------
__global__ void __launch_bounds__(256)
mv_meanx(const float* __restrict__ povs, const int* __restrict__ basep,
         float* __restrict__ meanXb) {
  int lane = threadIdx.x & 31;
  int wav  = threadIdx.x >> 5;
  int row  = blockIdx.x * 8 + wav;                   // 0..1023 = b*128+c
  const float* X = povs + (size_t)(*basep) * NB * CH * AREA + (size_t)row * AREA;
  float s = 0.f;
  for (int i = lane; i < AREA; i += 32) s += X[i];
  s += __shfl_xor(s, 1);
  s += __shfl_xor(s, 2);
  s += __shfl_xor(s, 4);
  s += __shfl_xor(s, 8);
  s += __shfl_xor(s, 16);
  if (lane == 0) meanXb[row] = s * (1.0f / AREA);
}

// ---------------------------------------------------------------------------
// Stage 2: fused flash attention.
// Block = 8 waves, all on the same (ov, b), 8 consecutive q-tiles.
// K/V tiles (32 keys) staged block-wide into LDS via async loads, double
// buffered.  Online softmax: row max via permlane16 packed-f16 butterflies,
// row sums via an extra WMMA against an all-ones B operand.
// Emits per-qtile column sums of the attention output.
// ---------------------------------------------------------------------------
__global__ void __launch_bounds__(256)
mv_attn(const _Float16* __restrict__ Qh, const _Float16* __restrict__ Kh,
        const _Float16* __restrict__ Vh, float* __restrict__ Zpart) {
  __shared__ __align__(16) _Float16 Klds[2][32 * 64]; // 2 x 4KB  [key][d]
  __shared__ __align__(16) _Float16 Vlds[2][64 * 32]; // 2 x 4KB  [d][key]
  __shared__ __align__(16) _Float16 Plds[8][16 * 32]; // per-wave P staging

  const int t    = threadIdx.x;
  const int lane = t & 31;
  const int wav  = t >> 5;
  const int m    = lane & 15;
  const int hh   = lane >> 4;

  const int g  = blockIdx.x;                         // 0..431
  const int vb = g / 18;                             // (ov, b) unit
  const int qt = (g % 18) * 8 + wav;                 // this wave's q-tile
  const int b  = vb & 7;
  const int ov = vb >> 3;

  // Q fragments for this 16-row tile (d = 0..31 and 32..63), kept resident.
  const _Float16* Qrow = Qh + ((size_t)b * AREA + (size_t)qt * 16 + m) * C2;
  const v16h qa0 = mv_frag(Qrow +  8 * hh,      Qrow + 16 +  8 * hh);
  const v16h qa1 = mv_frag(Qrow + 32 + 8 * hh,  Qrow + 48 +  8 * hh);

  const _Float16* Kbase = Kh + ((size_t)(ov * NB + b) * AREA) * C2;   // [key][d]
  const _Float16* Vbase = Vh + ((size_t)(ov * NB + b) * C2) * AREA;   // [d][key]

  // All-ones B operand: P @ ones gives per-row sums replicated into the
  // exact C-layout slots used by the per-lane row stats.
  v16h ones;
#pragma unroll
  for (int i = 0; i < 16; ++i) ones[i] = (_Float16)1.0f;

  // Per-lane row stats: element r <-> query row (r + 8*hh)  (matches C layout).
  float mrow[8], lrow[8];
  v8f o0 = {}, o1 = {}, o2 = {}, o3 = {};
#pragma unroll
  for (int r = 0; r < 8; ++r) { mrow[r] = -__builtin_inff(); lrow[r] = 0.f; }

  _Float16* P = &Plds[wav][0];

  // Cooperative tile staging: 256 threads x 16B cover each 4KB tile.
  const int krow = t >> 3, kc = t & 7;               // K tile: 32 rows x 128B
  const int vrow = t >> 2, vc = t & 3;               // V tile: 64 rows x 64B

  // prologue: stage tile 0 into buffer 0
  mv_async_ld16(mv_lds_off(&Klds[0][krow * 64 + kc * 8]),
                Kbase + (size_t)krow * C2 + kc * 8);
  mv_async_ld16(mv_lds_off(&Vlds[0][vrow * 32 + vc * 8]),
                Vbase + (size_t)vrow * AREA + vc * 8);

#pragma unroll 1
  for (int it = 0; it < NKIT; ++it) {
    const int cur = it & 1;
    if (it + 1 < NKIT) {                             // prefetch next tile
      const int kb1 = (it + 1) * 32;
      mv_async_ld16(mv_lds_off(&Klds[cur ^ 1][krow * 64 + kc * 8]),
                    Kbase + (size_t)(kb1 + krow) * C2 + kc * 8);
      mv_async_ld16(mv_lds_off(&Vlds[cur ^ 1][vrow * 32 + vc * 8]),
                    Vbase + (size_t)vrow * AREA + kb1 + vc * 8);
      asm volatile("s_wait_asynccnt 0x2" ::: "memory");  // current tile done
    } else {
      asm volatile("s_wait_asynccnt 0x0" ::: "memory");
    }
    __syncthreads();                                 // tile visible block-wide

    // ---- scores: S0 = keys 0..15 of tile, S1 = keys 16..31 ----
    v8f s0 = {}, s1 = {};
    {
      const _Float16* Kr = &Klds[cur][(0 * 16 + m) * 64];
      s0 = mv_wmma(qa0, mv_frag(Kr +      8 * hh, Kr + 16 + 8 * hh), s0);
      s0 = mv_wmma(qa1, mv_frag(Kr + 32 + 8 * hh, Kr + 48 + 8 * hh), s0);
    }
    {
      const _Float16* Kr = &Klds[cur][(1 * 16 + m) * 64];
      s1 = mv_wmma(qa0, mv_frag(Kr +      8 * hh, Kr + 16 + 8 * hh), s1);
      s1 = mv_wmma(qa1, mv_frag(Kr + 32 + 8 * hh, Kr + 48 + 8 * hh), s1);
    }

    // ---- row max of the 32 new columns: packed-f16 permlane butterflies ----
    float mnew[8], sc[8];
    {
      unsigned pk[4];
#pragma unroll
      for (int j = 0; j < 4; ++j) {
        h2 c2v;
        c2v[0] = (_Float16)fmaxf(s0[2 * j],     s1[2 * j]);
        c2v[1] = (_Float16)fmaxf(s0[2 * j + 1], s1[2 * j + 1]);
        unsigned v = __builtin_bit_cast(unsigned, c2v);
        v = mv_hmax2(v, mv_pl16<0x67452301u, 0xEFCDAB89u>(v));  // xor 1
        v = mv_hmax2(v, mv_pl16<0x54761032u, 0xDCFE98BAu>(v));  // xor 2
        v = mv_hmax2(v, mv_pl16<0x32107654u, 0xBA98FEDCu>(v));  // xor 4
        v = mv_hmax2(v, mv_pl16<0xFEDCBA98u, 0x76543210u>(v));  // xor 8
        pk[j] = v;
      }
#pragma unroll
      for (int r = 0; r < 8; ++r) {
        h2 v = __builtin_bit_cast(h2, pk[r >> 1]);
        float tmax = (float)v[r & 1];
        mnew[r] = fmaxf(mrow[r], tmax);
        sc[r]   = __expf(mrow[r] - mnew[r]);
        mrow[r] = mnew[r];
      }
    }

    // ---- P = exp(S - mnew): stage to LDS (C layout -> row-major [q][k]) ----
#pragma unroll
    for (int r = 0; r < 8; ++r) {
      float p0 = __expf(s0[r] - mnew[r]);
      float p1 = __expf(s1[r] - mnew[r]);
      P[(r + 8 * hh) * 32 + m]      = (_Float16)p0;
      P[(r + 8 * hh) * 32 + 16 + m] = (_Float16)p1;
      o0[r] *= sc[r]; o1[r] *= sc[r]; o2[r] *= sc[r]; o3[r] *= sc[r];
    }
    __syncthreads();                                 // P writes visible

    const _Float16* Prow = P + m * 32;
    v16h pa = mv_frag(Prow + 8 * hh, Prow + 16 + 8 * hh);

    // ---- row sums via WMMA against all-ones B ----
    v8f ls = {};
    ls = mv_wmma(pa, ones, ls);
#pragma unroll
    for (int r = 0; r < 8; ++r) lrow[r] = lrow[r] * sc[r] + ls[r];

    // ---- O += P @ V  (4 d-subtiles of 16) ----
    {
      const _Float16* Vr = &Vlds[cur][(0 * 16 + m) * 32];
      o0 = mv_wmma(pa, mv_frag(Vr + 8 * hh, Vr + 16 + 8 * hh), o0);
    }
    {
      const _Float16* Vr = &Vlds[cur][(1 * 16 + m) * 32];
      o1 = mv_wmma(pa, mv_frag(Vr + 8 * hh, Vr + 16 + 8 * hh), o1);
    }
    {
      const _Float16* Vr = &Vlds[cur][(2 * 16 + m) * 32];
      o2 = mv_wmma(pa, mv_frag(Vr + 8 * hh, Vr + 16 + 8 * hh), o2);
    }
    {
      const _Float16* Vr = &Vlds[cur][(3 * 16 + m) * 32];
      o3 = mv_wmma(pa, mv_frag(Vr + 8 * hh, Vr + 16 + 8 * hh), o3);
    }
    __syncthreads();                                 // tile reads done
  }

  // ---- finalize: column sums over the 16 query rows of O / l ----
  float il[8];
#pragma unroll
  for (int r = 0; r < 8; ++r) il[r] = 1.0f / lrow[r];

  float* zout = Zpart + ((size_t)(ov * NB + b) * NQT + qt) * 64;
  {
    float cs = 0.f;
#pragma unroll
    for (int r = 0; r < 8; ++r) cs += o0[r] * il[r];
    cs += __shfl_xor(cs, 16);
    if (lane < 16) zout[0 * 16 + m] = cs;
  }
  {
    float cs = 0.f;
#pragma unroll
    for (int r = 0; r < 8; ++r) cs += o1[r] * il[r];
    cs += __shfl_xor(cs, 16);
    if (lane < 16) zout[1 * 16 + m] = cs;
  }
  {
    float cs = 0.f;
#pragma unroll
    for (int r = 0; r < 8; ++r) cs += o2[r] * il[r];
    cs += __shfl_xor(cs, 16);
    if (lane < 16) zout[2 * 16 + m] = cs;
  }
  {
    float cs = 0.f;
#pragma unroll
    for (int r = 0; r < 8; ++r) cs += o3[r] * il[r];
    cs += __shfl_xor(cs, 16);
    if (lane < 16) zout[3 * 16 + m] = cs;
  }
}

// ---------------------------------------------------------------------------
// Stage 3a: deterministic reduction of per-qtile partials.
// ---------------------------------------------------------------------------
__global__ void __launch_bounds__(256)
mv_reduce_z(const float* __restrict__ Zpart, float* __restrict__ Zred) {
  int tt = blockIdx.x * 256 + threadIdx.x;           // 0..1535
  if (tt >= 3 * NB * 64) return;
  int d  = tt & 63;
  int vb = tt >> 6;
  const float* p = Zpart + (size_t)vb * NQT * 64 + d;
  float s = 0.f;
  for (int q = 0; q < NQT; ++q) s += p[(size_t)q * 64];
  Zred[tt] = s;
}

// ---------------------------------------------------------------------------
// Stage 3b: epilogue GEMMs -> out [8,128] f32.
// ---------------------------------------------------------------------------
__global__ void __launch_bounds__(1024)
mv_finalize(const float* __restrict__ Zred, const float* __restrict__ meanXb,
            const float* __restrict__ Wz, const float* __restrict__ bz,
            const float* __restrict__ Wfc, const float* __restrict__ bfc,
            float* __restrict__ out) {
  __shared__ float pooled[NB * CH];
  int tt = threadIdx.x;
  int b = tt >> 7;
  int c = tt & 127;

  float acc = meanXb[b * CH + c] + 3.0f * bz[c];
#pragma unroll 8
  for (int d = 0; d < C2; ++d) {
    float z = (Zred[(0 * NB + b) * 64 + d] +
               Zred[(1 * NB + b) * 64 + d] +
               Zred[(2 * NB + b) * 64 + d]) * (1.0f / AREA);
    acc += Wz[c * C2 + d] * z;
  }
  pooled[tt] = acc * 0.25f;
  __syncthreads();

  float r = bfc[c];                                  // c plays role of output o
#pragma unroll 8
  for (int cc = 0; cc < CH; ++cc) r += Wfc[c * CH + cc] * pooled[b * CH + cc];
  out[tt] = r;
}

// ---------------------------------------------------------------------------
// Launch.
// ---------------------------------------------------------------------------
extern "C" void kernel_launch(void* const* d_in, const int* in_sizes, int n_in,
                              void* d_out, int out_size, void* d_ws, size_t ws_size,
                              hipStream_t stream) {
  const float* povs = (const float*)d_in[0];
  const float* Wq   = (const float*)d_in[1];
  const float* bq   = (const float*)d_in[2];
  const float* Wk   = (const float*)d_in[3];
  const float* bk   = (const float*)d_in[4];
  const float* Wv   = (const float*)d_in[5];
  const float* bv   = (const float*)d_in[6];
  const float* Wz   = (const float*)d_in[7];
  const float* bz   = (const float*)d_in[8];
  const float* Wfc  = (const float*)d_in[9];
  const float* bfc  = (const float*)d_in[10];
  const int*   basep = (const int*)d_in[11];
  float* out = (float*)d_out;

  // Workspace layout (bytes)
  char* ws = (char*)d_ws;
  _Float16* Qh = (_Float16*)(ws + 0);                     // 8*2304*64*2   = 2,359,296
  _Float16* Kh = (_Float16*)(ws + 2359296);               // 3*8*2304*64*2 = 7,077,888
  _Float16* Vh = (_Float16*)(ws + 9437184);               // 3*8*64*2304*2 = 7,077,888
  float* Zpart  = (float*)(ws + 16515072);                // 3*8*144*64*4  =   884,736
  float* Zred   = (float*)(ws + 17399808);                // 1536*4        =     6,144
  float* meanXb = (float*)(ws + 17405952);                // 1024*4        =     4,096
  (void)in_sizes; (void)n_in; (void)out_size; (void)ws_size;

  // Stage 1: projections (f16) + base-view spatial means
  mv_proj_q<<<dim3((NB * AREA * 64) / 256), dim3(256), 0, stream>>>(povs, Wq, bq, basep, Qh);
  mv_proj_k<<<dim3((3 * NB * AREA * 64) / 256), dim3(256), 0, stream>>>(povs, Wk, bk, basep, Kh);
  mv_proj_v<<<dim3((3 * NB * AREA * 64) / 256), dim3(256), 0, stream>>>(povs, Wv, bv, basep, Vh);
  mv_meanx <<<dim3((NB * CH) / 8), dim3(256), 0, stream>>>(povs, basep, meanXb);

  // Stage 2: fused flash attention (WMMA + async tile staging).
  mv_attn<<<dim3(432), dim3(256), 0, stream>>>(Qh, Kh, Vh, Zpart);

  // Stage 3: deterministic reductions + epilogue GEMMs
  mv_reduce_z<<<dim3(6), dim3(256), 0, stream>>>(Zpart, Zred);
  mv_finalize<<<dim3(1), dim3(1024), 0, stream>>>(Zred, meanXb, Wz, bz, Wfc, bfc, out);
}
